// Model_29300266893901
// MI455X (gfx1250) — compile-verified, matrix operand order checked by reading, not run on hardware
//
#include <hip/hip_runtime.h>

// ---------------- CDNA5 WMMA types ----------------
typedef __attribute__((ext_vector_type(16))) __bf16 v16bf;
typedef __attribute__((ext_vector_type(8)))  float  v8f;

union FragU { v16bf v; unsigned u[8]; };

__device__ __forceinline__ unsigned pack_bf16(float lo, float hi) {
    unsigned a = __builtin_bit_cast(unsigned, lo);
    unsigned b = __builtin_bit_cast(unsigned, hi);
    // round-to-nearest-even truncation f32 -> bf16
    a = (a + 0x7FFFu + ((a >> 16) & 1u)) >> 16;
    b = (b + 0x7FFFu + ((b >> 16) & 1u)) >> 16;
    return (b << 16) | (a & 0xFFFFu);
}

#define NODES 20000
#define SAXIS 2
#define MROWS (NODES * SAXIS)   // 40000
#define HDIM  246
#define HPAD  256
#define NT    16                // 256/16 column tiles
#define NGRP  4                 // N-tiles per wave (64 columns)

// ---------------- degrees & norms ----------------
__global__ void deg_kernel(const int* __restrict__ src, const int* __restrict__ dst,
                           float* __restrict__ deg_out, float* __restrict__ deg_in, int E) {
    int i = blockIdx.x * blockDim.x + threadIdx.x;
    if (i >= E) return;
    atomicAdd(&deg_out[src[i]], 1.0f);
    atomicAdd(&deg_in[dst[i]], 1.0f);
}

__global__ void norm_kernel(const float* __restrict__ deg_out, const float* __restrict__ deg_in,
                            float* __restrict__ norm_out, float* __restrict__ norm_in, int n) {
    int i = blockIdx.x * blockDim.x + threadIdx.x;
    if (i >= n) return;
    norm_out[i] = rsqrtf(fmaxf(deg_out[i], 1.0f));
    norm_in[i]  = rsqrtf(fmaxf(deg_in[i], 1.0f));
}

// ---------------- pack W [K x Hd] (f32 row-major) into WMMA B-fragment layout ----------------
// Per (ktile, ntile) tile: 32 lanes x 8 dwords (packed bf16 pairs), matching the
// 16-bit 32x16 B operand layout (lane<16: K 0-7 / 16-23; lane>=16: K 8-15 / 24-31).
__global__ void pack_w_kernel(const float* __restrict__ W, int K, int Hd, int ktiles,
                              unsigned* __restrict__ out) {
    int gid = blockIdx.x * blockDim.x + threadIdx.x;
    int total = ktiles * NT * 32;
    if (gid >= total) return;
    int lane    = gid & 31;
    int tileIdx = gid >> 5;
    int kt = tileIdx >> 4;          // / NT
    int tn = tileIdx & (NT - 1);
    int col  = tn * 16 + (lane & 15);
    int hi16 = lane >> 4;
    unsigned* o = out + (size_t)tileIdx * 32 * 8 + (size_t)lane * 8;
#pragma unroll
    for (int j = 0; j < 8; ++j) {
        int base = (j < 4) ? (hi16 ? 8 + 2 * j : 2 * j)
                           : (hi16 ? 24 + 2 * (j - 4) : 16 + 2 * (j - 4));
        int k0 = kt * 32 + base;
        float lo = (k0     < K && col < Hd) ? W[(size_t)k0 * Hd + col]       : 0.0f;
        float hi = (k0 + 1 < K && col < Hd) ? W[(size_t)(k0 + 1) * Hd + col] : 0.0f;
        o[j] = pack_bf16(lo, hi);
    }
}

// ---------------- edge aggregation: agg[dst] += X[src] * norm_out[src] ----------------
// blockFloats = S * rowWidth (floats per node). One float4 chunk per thread.
__global__ void agg_kernel(const float* __restrict__ X, const int* __restrict__ src,
                           const int* __restrict__ dst, const float* __restrict__ norm_out,
                           float* __restrict__ agg, int blockFloats, int E) {
    long long idx = (long long)blockIdx.x * blockDim.x + threadIdx.x;
    int cpe = blockFloats >> 2;
    long long total = (long long)E * cpe;
    if (idx >= total) return;
    int e = (int)(idx / cpe);
    int c = (int)(idx % cpe);
    int s = src[e], d = dst[e];
    float nw = norm_out[s];
    const float4 x = *(const float4*)(X + (size_t)s * blockFloats + (size_t)c * 4);
    float* o = agg + (size_t)d * blockFloats + (size_t)c * 4;
    atomicAdd(o + 0, x.x * nw);
    atomicAdd(o + 1, x.y * nw);
    atomicAdd(o + 2, x.z * nw);
    atomicAdd(o + 3, x.w * nw);
}

// ---------------- GEMM: Y[M x HPAD] = relu( (X*norm_in) @ Wpack + bias ) ----------------
// 8 waves/block; each wave computes 4 adjacent 16x16 tiles (one A fragment reused
// across 4 v_wmma_f32_16x16x32_bf16). Full packed-W panel staged in LDS once per
// block. Per k-step: issue global A loads, then all 8 ds_load_b128 for the 4 B
// fragments, then do the bf16 pack ALU work (hides both latencies), then 4
// back-to-back WMMAs behind a single dscnt wait.
__global__ void gcn_mm_kernel(const float* __restrict__ X, int Kp,
                              const unsigned* __restrict__ Wp,
                              const float* __restrict__ bias,
                              const float* __restrict__ norm_in,
                              float* __restrict__ Y) {
    extern __shared__ unsigned lds[];
    const int ktiles = Kp >> 5;

    // ---- cooperative stage of the whole packed-W panel into LDS ----
    {
        const int totalVec = ktiles * NT * 32 * 8 / 4;   // uint4 count
        const uint4* gsrc = (const uint4*)Wp;
        uint4* lsb = (uint4*)lds;
        for (int i = threadIdx.x; i < totalVec; i += 256)
            lsb[i] = gsrc[i];
    }
    __syncthreads();

    int wave = threadIdx.x >> 5;
    int lane = threadIdx.x & 31;
    int tile = blockIdx.x * (blockDim.x >> 5) + wave;   // exact grid -> EXEC all ones
    int tm   = tile >> 2;        // M tile
    int g    = tile & 3;         // column group (4 N-tiles = 64 cols)
    int half = lane >> 4;        // 0: K 0-7/16-23   1: K 8-15/24-31
    int l15  = lane & 15;

    int row = tm * 16 + l15;                    // A-matrix row for this lane
    float scale = norm_in[row >> 1];            // node = row / S  (S == 2)
    const float* xr = X + (size_t)row * Kp;
    // this lane's base within a B tile: lane*32 bytes (8 dwords)
    const unsigned* lb = lds + ((size_t)(g * NGRP) * 32 + lane) * 8;

    v8f acc[NGRP];
#pragma unroll
    for (int t = 0; t < NGRP; ++t) acc[t] = (v8f){};

    for (int kt = 0; kt < ktiles; ++kt) {
        int kb = kt * 32 + half * 8;
        // prefetch next k-step's A line (global_prefetch_b8; speculative, safe at end)
        __builtin_prefetch(xr + kb + 32, 0, 3);

        // ---- issue global A loads first (VMEM in flight) ----
        float4 x0 = *(const float4*)(xr + kb);
        float4 x1 = *(const float4*)(xr + kb + 4);
        float4 x2 = *(const float4*)(xr + kb + 16);
        float4 x3 = *(const float4*)(xr + kb + 20);

        // ---- issue all B-fragment LDS loads (DS in flight) ----
        const unsigned* bkt = lb + (size_t)kt * NT * 32 * 8;
        uint4 blo[NGRP], bhi[NGRP];
#pragma unroll
        for (int t = 0; t < NGRP; ++t) {
            const unsigned* bp = bkt + (size_t)t * 32 * 8;
            blo[t] = *(const uint4*)(bp);
            bhi[t] = *(const uint4*)(bp + 4);
        }

        // ---- ALU: bf16 pack of the A fragment (hides VMEM/DS latency) ----
        FragU a;
        a.u[0] = pack_bf16(x0.x * scale, x0.y * scale);
        a.u[1] = pack_bf16(x0.z * scale, x0.w * scale);
        a.u[2] = pack_bf16(x1.x * scale, x1.y * scale);
        a.u[3] = pack_bf16(x1.z * scale, x1.w * scale);
        a.u[4] = pack_bf16(x2.x * scale, x2.y * scale);
        a.u[5] = pack_bf16(x2.z * scale, x2.w * scale);
        a.u[6] = pack_bf16(x3.x * scale, x3.y * scale);
        a.u[7] = pack_bf16(x3.z * scale, x3.w * scale);

        // ---- 4 back-to-back WMMAs ----
#pragma unroll
        for (int t = 0; t < NGRP; ++t) {
            FragU b;
            b.u[0] = blo[t].x; b.u[1] = blo[t].y; b.u[2] = blo[t].z; b.u[3] = blo[t].w;
            b.u[4] = bhi[t].x; b.u[5] = bhi[t].y; b.u[6] = bhi[t].z; b.u[7] = bhi[t].w;
            acc[t] = __builtin_amdgcn_wmma_f32_16x16x32_bf16(
                /*neg_a=*/false, a.v, /*neg_b=*/false, b.v,
                /*c_mod=*/(short)0, acc[t], /*reuse_a=*/false, /*reuse_b=*/false);
        }
    }

    int rbase = tm * 16 + half * 8;
#pragma unroll
    for (int t = 0; t < NGRP; ++t) {
        int col = (g * NGRP + t) * 16 + l15;
        float bv = (col < HDIM) ? bias[col] : 0.0f;   // pad columns -> relu(0) = 0
#pragma unroll
        for (int r = 0; r < 8; ++r) {
            float y = acc[t][r] + bv;
            y = y > 0.0f ? y : 0.0f;
            Y[(size_t)(rbase + r) * HPAD + col] = y;
        }
    }
}

// ---------------- final: out[n] = mean_s(h[n,s,:]) . fc_w + fc_b ----------------
__global__ void final_kernel(const float* __restrict__ h, const float* __restrict__ fw,
                             const float* __restrict__ fb, float* __restrict__ out, int n) {
    int i = blockIdx.x * blockDim.x + threadIdx.x;
    if (i >= n) return;
    const float* p = h + (size_t)i * SAXIS * HPAD;
    float acc = 0.0f;
    for (int f = 0; f < HDIM; ++f)
        acc += (p[f] + p[HPAD + f]) * fw[f];
    out[i] = acc * (1.0f / SAXIS) + fb[0];
}

// ---------------- launcher ----------------
extern "C" void kernel_launch(void* const* d_in, const int* in_sizes, int n_in,
                              void* d_out, int out_size, void* d_ws, size_t ws_size,
                              hipStream_t stream) {
    const float* feat = (const float*)d_in[0];   // [N, S, 128]
    const int*   src  = (const int*)d_in[1];
    const int*   dst  = (const int*)d_in[2];
    const float* W0   = (const float*)d_in[3];
    const float* b0   = (const float*)d_in[4];
    const float* W1   = (const float*)d_in[5];
    const float* b1   = (const float*)d_in[6];
    const float* W2   = (const float*)d_in[7];
    const float* b2   = (const float*)d_in[8];
    const float* fcw  = (const float*)d_in[9];
    const float* fcb  = (const float*)d_in[10];
    float* out = (float*)d_out;
    const int E = in_sizes[1];

    // workspace carve-up
    char* w = (char*)d_ws;
    size_t off = 0;
    auto take = [&](size_t bytes) -> char* {
        char* p = w + off;
        off += (bytes + 255) & ~(size_t)255;
        return p;
    };
    float*    deg_out  = (float*)take((size_t)NODES * 4);
    float*    deg_in   = (float*)take((size_t)NODES * 4);
    float*    norm_out = (float*)take((size_t)NODES * 4);
    float*    norm_in  = (float*)take((size_t)NODES * 4);
    unsigned* packW0   = (unsigned*)take((size_t)4 * NT * 32 * 8 * 4);   // K=128 -> 4 ktiles
    unsigned* packW1   = (unsigned*)take((size_t)8 * NT * 32 * 8 * 4);   // Kp=256 -> 8 ktiles
    unsigned* packW2   = (unsigned*)take((size_t)8 * NT * 32 * 8 * 4);
    float*    bufA     = (float*)take((size_t)MROWS * HPAD * 4);
    float*    bufB     = (float*)take((size_t)MROWS * HPAD * 4);
    float*    aggB     = (float*)take((size_t)MROWS * HPAD * 4);
    (void)ws_size; (void)n_in; (void)out_size;

    const int T = 256;
    // degrees + norms
    hipMemsetAsync(deg_out, 0, (size_t)NODES * 4, stream);
    hipMemsetAsync(deg_in,  0, (size_t)NODES * 4, stream);
    deg_kernel<<<(E + T - 1) / T, T, 0, stream>>>(src, dst, deg_out, deg_in, E);
    norm_kernel<<<(NODES + T - 1) / T, T, 0, stream>>>(deg_out, deg_in, norm_out, norm_in, NODES);

    // pack weights into WMMA B-fragment layout
    pack_w_kernel<<<(4 * NT * 32 + T - 1) / T, T, 0, stream>>>(W0, 128,  HDIM, 4, packW0);
    pack_w_kernel<<<(8 * NT * 32 + T - 1) / T, T, 0, stream>>>(W1, HDIM, HDIM, 8, packW1);
    pack_w_kernel<<<(8 * NT * 32 + T - 1) / T, T, 0, stream>>>(W2, HDIM, HDIM, 8, packW2);

    // 2500 M-tiles x 4 column-groups = 10000 waves = 1250 blocks (exact, no guards)
    const int mmBlocks = (MROWS / 16) * (NT / NGRP) / 8;
    const size_t ldsBytes128 = (size_t)4 * NT * 32 * 8 * 4;   // 64 KB  (Kp=128)
    const size_t ldsBytes256 = (size_t)8 * NT * 32 * 8 * 4;   // 128 KB (Kp=256)

    // ---- layer 0: K = 128 (input features, contiguous [N,S,128]) ----
    {
        int bf = SAXIS * 128;                          // 256 floats per node
        long long tot = (long long)E * (bf >> 2);
        hipMemsetAsync(aggB, 0, (size_t)MROWS * 128 * 4, stream);
        agg_kernel<<<(unsigned)((tot + T - 1) / T), T, 0, stream>>>(feat, src, dst, norm_out, aggB, bf, E);
        gcn_mm_kernel<<<mmBlocks, T, ldsBytes128, stream>>>(aggB, 128, packW0, b0, norm_in, bufA);
    }
    // ---- layer 1: K = 256 (padded, pad cols are zero) ----
    {
        int bf = SAXIS * HPAD;                         // 512 floats per node
        long long tot = (long long)E * (bf >> 2);
        hipMemsetAsync(aggB, 0, (size_t)MROWS * HPAD * 4, stream);
        agg_kernel<<<(unsigned)((tot + T - 1) / T), T, 0, stream>>>(bufA, src, dst, norm_out, aggB, bf, E);
        gcn_mm_kernel<<<mmBlocks, T, ldsBytes256, stream>>>(aggB, HPAD, packW1, b1, norm_in, bufB);
    }
    // ---- layer 2 ----
    {
        int bf = SAXIS * HPAD;
        long long tot = (long long)E * (bf >> 2);
        hipMemsetAsync(aggB, 0, (size_t)MROWS * HPAD * 4, stream);
        agg_kernel<<<(unsigned)((tot + T - 1) / T), T, 0, stream>>>(bufB, src, dst, norm_out, aggB, bf, E);
        gcn_mm_kernel<<<mmBlocks, T, ldsBytes256, stream>>>(aggB, HPAD, packW2, b2, norm_in, bufA);
    }

    // ---- head: mean over S, dot fc_w ----
    final_kernel<<<(NODES + T - 1) / T, T, 0, stream>>>(bufA, fcw, fcb, out, NODES);
}
